// MAEModel_71708773974903
// MI455X (gfx1250) — matrix operand order repaired and checked by reading, not compiled, not run on hardware
//
#include <hip/hip_runtime.h>
#include <hip/hip_bf16.h>

// ---------------------------------------------------------------------------
// CDNA5 (gfx1250) DeeperGCN MAE forward.
//   - Dense projections / MLPs on v_wmma_f32_16x16x32_f16: block tile 128x32,
//     K-chunk 64, 8 waves, each wave owns a 16x32 strip (2 accumulators,
//     4 WMMA per chunk). A/B staged in LDS *in fragment order* so each
//     WMMA operand is one contiguous 32B LDS read (ds_load_b128 pairs).
//     Staging has a block-uniform guard-free fast path (full tiles) and a
//     guarded slow path for tails / tiny GEMMs.
//   - Scatter-softmax aggregation: relu(m)+1e-7 > 0  =>  segment-max via
//     integer atomicMax on float bits; exp/num/den via global f32 atomics.
//   - Graph-capture safe: kernels + hipMemsetAsync only, all on `stream`.
// ---------------------------------------------------------------------------

typedef __attribute__((ext_vector_type(16))) _Float16 v16h;
typedef __attribute__((ext_vector_type(4)))  _Float16 v4h;
typedef __attribute__((ext_vector_type(8)))  float    v8f;

#define GEMM_TM 128
#define GEMM_TN 32
#define GEMM_TK 64

// C[M,N] = act( A[M,K] @ W[K,N] + bias (+ res) )
__global__ __launch_bounds__(256) void gemm_wmma_f16(
    const float* __restrict__ A, const float* __restrict__ W,
    const float* __restrict__ bias, const float* __restrict__ res,
    float* __restrict__ C, int M, int K, int N, int relu)
{
  // Fragment-order staging (contiguous k = 16*(lane/16) + e per 32-k slice):
  //  A: [rb(8)][kc(2)][lane(32)][e(16)] halves  -> 16 KB
  //  B: [kc(2)][nt(2)][lane(32)][e(16)] halves  ->  4 KB
  __shared__ _Float16 sA[8 * 2 * 32 * 16];
  __shared__ _Float16 sB[2 * 2 * 32 * 16];

  const int tid  = threadIdx.x;
  const int wave = tid >> 5;
  const int lane = tid & 31;
  const int rowBase = blockIdx.x * GEMM_TM;
  const int colBase = blockIdx.y * GEMM_TN;

  // Block-uniform fast-path eligibility (scalar branches in the hot loop).
  const bool kvec   = (K & 3) == 0;
  const bool fullMA = kvec && (rowBase + GEMM_TM <= M);
  const bool fullNB = ((N & 3) == 0) && (colBase + GEMM_TN <= N);

  v8f acc0 = {}, acc1 = {};

  for (int k0 = 0; k0 < K; k0 += GEMM_TK) {
    const bool fullK = (k0 + GEMM_TK <= K);

    // Prefetch next A chunk (global_prefetch_b8) while we stage this one.
    if (k0 + GEMM_TK < K) {
      int pr = rowBase + (tid >> 1);
      if (pr < M) __builtin_prefetch(A + (size_t)pr * K + k0 + GEMM_TK, 0, 1);
    }

    // ---- stage A (128 x 64): 8 x float4 per thread, scattered to fragment
    //      order. A float4 (k aligned 4) always lands in one 8-half run, so
    //      the LDS store is a single contiguous v4h (ds_store_b64).
    if (fullMA && fullK) {
      // guard-free: pure global_load_b128 -> ds_store_b64
      #pragma unroll
      for (int t = 0; t < 8; ++t) {
        int id = tid + t * 256;          // vec4 id in [0, 128*16)
        int r  = id >> 4;
        int kl = (id & 15) << 2;         // 0..60
        float4 v = *reinterpret_cast<const float4*>(
            A + (size_t)(rowBase + r) * K + k0 + kl);
        int kk = kl & 31, kc = kl >> 5;
        int ln = (r & 15) + ((kk >= 16) ? 16 : 0);
        int e  = kk & 15;
        int base = ((((r >> 4) * 2 + kc) * 32 + ln) << 4) + e;
        v4h hv = { (_Float16)v.x, (_Float16)v.y, (_Float16)v.z, (_Float16)v.w };
        *reinterpret_cast<v4h*>(&sA[base]) = hv;
      }
    } else {
      #pragma unroll
      for (int t = 0; t < 8; ++t) {
        int id = tid + t * 256;
        int r  = id >> 4;
        int kl = (id & 15) << 2;
        int gr = rowBase + r, gc = k0 + kl;
        float4 v = make_float4(0.f, 0.f, 0.f, 0.f);
        if (gr < M) {
          if (kvec && (gc + 3) < K) {
            v = *reinterpret_cast<const float4*>(A + (size_t)gr * K + gc);
          } else {
            const float* ap = A + (size_t)gr * K;
            if (gc + 0 < K) v.x = ap[gc + 0];
            if (gc + 1 < K) v.y = ap[gc + 1];
            if (gc + 2 < K) v.z = ap[gc + 2];
            if (gc + 3 < K) v.w = ap[gc + 3];
          }
        }
        int kk = kl & 31, kc = kl >> 5;
        int ln = (r & 15) + ((kk >= 16) ? 16 : 0);
        int e  = kk & 15;
        int base = ((((r >> 4) * 2 + kc) * 32 + ln) << 4) + e;
        v4h hv = { (_Float16)v.x, (_Float16)v.y, (_Float16)v.z, (_Float16)v.w };
        *reinterpret_cast<v4h*>(&sA[base]) = hv;
      }
    }

    // ---- stage B (64 x 32): fast path = 2 x float4 per thread, scattered
    //      (4 ds_store_b16 each at 32B stride); slow path = 8 guarded scalars.
    if (fullNB && fullK) {
      #pragma unroll
      for (int t = 0; t < 2; ++t) {
        int id = tid + t * 256;          // vec4 id over 64*8
        int kl = id >> 3;
        int n4 = (id & 7) << 2;
        float4 v = *reinterpret_cast<const float4*>(
            W + (size_t)(k0 + kl) * N + colBase + n4);
        int kk = kl & 31, kc = kl >> 5;
        int lnB = (kk >= 16) ? 16 : 0;
        int e   = kk & 15;
        _Float16 hv[4] = { (_Float16)v.x, (_Float16)v.y,
                           (_Float16)v.z, (_Float16)v.w };
        #pragma unroll
        for (int j = 0; j < 4; ++j) {
          int nl = n4 + j;
          int ln = (nl & 15) + lnB;
          sB[(((kc * 2 + (nl >> 4)) * 32 + ln) << 4) + e] = hv[j];
        }
      }
    } else {
      #pragma unroll
      for (int t = 0; t < 8; ++t) {
        int id = tid + t * 256;
        int kl = id >> 5, nl = id & 31;
        int gr = k0 + kl, gc = colBase + nl;
        float v = (gr < K && gc < N) ? W[(size_t)gr * N + gc] : 0.f;
        int kk = kl & 31, kc = kl >> 5;
        int ln = (nl & 15) + ((kk >= 16) ? 16 : 0);
        int off = (((kc * 2 + (nl >> 4)) * 32 + ln) << 4) + (kk & 15);
        sB[off] = (_Float16)v;
      }
    }
    __syncthreads();

    // ---- 4 WMMAs per chunk: one contiguous 32B LDS read per operand
    #pragma unroll
    for (int kc = 0; kc < 2; ++kc) {
      v16h af = *reinterpret_cast<const v16h*>(
          &sA[(((wave * 2 + kc) * 32 + lane) << 4)]);
      v16h bf0 = *reinterpret_cast<const v16h*>(
          &sB[(((kc * 2 + 0) * 32 + lane) << 4)]);
      v16h bf1 = *reinterpret_cast<const v16h*>(
          &sB[(((kc * 2 + 1) * 32 + lane) << 4)]);
      acc0 = __builtin_amdgcn_wmma_f32_16x16x32_f16(
          false, af, false, bf0, (short)0, acc0, false, false);
      acc1 = __builtin_amdgcn_wmma_f32_16x16x32_f16(
          false, af, false, bf1, (short)0, acc1, false, false);
    }
    __syncthreads();
  }

  // Epilogue: C layout — VGPR r holds M=r (lanes 0-15) / M=r+8 (lanes 16-31)
  const int mOff = (lane >= 16) ? 8 : 0;
  #pragma unroll
  for (int nt = 0; nt < 2; ++nt) {
    int gCol = colBase + nt * 16 + (lane & 15);
    if (gCol >= N) continue;
    float bv = bias ? bias[gCol] : 0.f;
    const v8f a = nt ? acc1 : acc0;
    #pragma unroll
    for (int r = 0; r < 8; ++r) {
      int gRow = rowBase + (wave << 4) + r + mOff;
      if (gRow < M) {
        size_t o = (size_t)gRow * N + gCol;
        float v = a[r] + bv;
        if (res)  v += res[o];
        if (relu) v = fmaxf(v, 0.0f);
        C[o] = v;
      }
    }
  }
}

// ------------------------- LayerNorm (wave per row) -------------------------
__global__ __launch_bounds__(256) void layernorm_rows(
    const float* __restrict__ X, const float* __restrict__ g,
    const float* __restrict__ b, float* __restrict__ Y, int n, int d)
{
  int wave = threadIdx.x >> 5, lane = threadIdx.x & 31;
  int row = blockIdx.x * 8 + wave;
  if (row >= n) return;
  const float* x = X + (size_t)row * d;
  float loc[4];                 // d <= 128
  int cnt = 0;
  float s = 0.f;
  for (int i = lane; i < d; i += 32) { float v = x[i]; loc[cnt++] = v; s += v; }
  #pragma unroll
  for (int off = 16; off; off >>= 1) s += __shfl_xor(s, off, 32);
  float mu = s / (float)d;
  float vs = 0.f;
  for (int c = 0; c < cnt; ++c) { float t = loc[c] - mu; vs += t * t; }
  #pragma unroll
  for (int off = 16; off; off >>= 1) vs += __shfl_xor(vs, off, 32);
  float rstd = rsqrtf(vs / (float)d + 1e-5f);
  float* y = Y + (size_t)row * d;
  int c2 = 0;
  for (int i = lane; i < d; i += 32) {
    y[i] = (loc[c2++] - mu) * rstd * g[i] + b[i];
  }
}

// ----------------------- masking (deterministic hash) -----------------------
__device__ __forceinline__ unsigned hash_u32(unsigned x) {
  x ^= x >> 16; x *= 0x7feb352dU; x ^= x >> 15; x *= 0x846ca68bU; x ^= x >> 16;
  return x;
}

__global__ void copy_mask_rows(const float* __restrict__ in, float* __restrict__ out,
                               int n, int d, unsigned seed, unsigned thresh)
{
  long long i = (long long)blockIdx.x * blockDim.x + threadIdx.x;
  if (i >= (long long)n * d) return;
  int row = (int)(i / d);
  float v = in[i];
  if (hash_u32((unsigned)row * 0x9E3779B9u + seed) < thresh) v = 0.0f;
  out[i] = v;
}

// -------------------- softmax-aggregation edge kernels ----------------------
// m = relu(hn[src] (+e)) + 1e-7 ; mmax[dst] = max (positive => int-bit order)
__global__ void edge_gather_relu_max(
    const float* __restrict__ hn, const float* __restrict__ eattr,
    const int* __restrict__ src, const int* __restrict__ dst,
    float* __restrict__ m, float* __restrict__ mmax, int E, int d)
{
  long long i = (long long)blockIdx.x * blockDim.x + threadIdx.x;
  if (i >= (long long)E * d) return;
  int e = (int)(i / d);
  int f = (int)(i - (long long)e * d);
  float v = hn[(size_t)src[e] * d + f];
  if (eattr) v += eattr[i];
  v = fmaxf(v, 0.0f) + 1e-7f;
  m[i] = v;
  atomicMax((int*)(mmax + (size_t)dst[e] * d + f), __float_as_int(v));
}

__global__ void edge_exp_scatter(
    const float* __restrict__ m, const float* __restrict__ mmax,
    const int* __restrict__ dst, float* __restrict__ num,
    float* __restrict__ den, int E, int d)
{
  long long i = (long long)blockIdx.x * blockDim.x + threadIdx.x;
  if (i >= (long long)E * d) return;
  int e = (int)(i / d);
  int f = (int)(i - (long long)e * d);
  size_t o = (size_t)dst[e] * d + f;
  float mv = m[i];
  float ex = __expf(mv - mmax[o]);
  atomicAdd(num + o, ex * mv);
  atomicAdd(den + o, ex);
}

__global__ void agg_finalize(const float* __restrict__ num,
                             const float* __restrict__ den,
                             float* __restrict__ agg, long long total)
{
  long long i = (long long)blockIdx.x * blockDim.x + threadIdx.x;
  if (i >= total) return;
  agg[i] = num[i] / (den[i] + 1e-16f);
}

// ------------------------------ host helpers --------------------------------
static inline void run_gemm(const float* A, const float* W, const float* bias,
                            const float* res, float* C, int M, int K, int N,
                            int relu, hipStream_t s)
{
  dim3 g((unsigned)((M + GEMM_TM - 1) / GEMM_TM),
         (unsigned)((N + GEMM_TN - 1) / GEMM_TN));
  gemm_wmma_f16<<<g, 256, 0, s>>>(A, W, bias, res, C, M, K, N, relu);
}

// One pre-norm DeeperGCN block. Writes the post-residual result into outC
// (outC == h for in-place layers; decoder final layer writes into d_out).
static void gen_layer(float* h, float* hn, float* mid, float* agg,
                      float* mmax, float* num, float* den, float* m,
                      const int* src, const int* dst, const float* eattr,
                      const float* W1, const float* b1,
                      const float* W2, const float* b2,
                      const float* ln_g, const float* ln_b,
                      int n, int E, int d, float* outC, hipStream_t s)
{
  layernorm_rows<<<(n + 7) / 8, 256, 0, s>>>(h, ln_g, ln_b, hn, n, d);
  size_t nd_bytes = (size_t)n * d * sizeof(float);
  hipMemsetAsync(mmax, 0, nd_bytes, s);
  hipMemsetAsync(num,  0, nd_bytes, s);
  hipMemsetAsync(den,  0, nd_bytes, s);
  long long tE = (long long)E * d;
  int be = (int)((tE + 255) / 256);
  edge_gather_relu_max<<<be, 256, 0, s>>>(hn, eattr, src, dst, m, mmax, E, d);
  edge_exp_scatter<<<be, 256, 0, s>>>(m, mmax, dst, num, den, E, d);
  long long tN = (long long)n * d;
  agg_finalize<<<(int)((tN + 255) / 256), 256, 0, s>>>(num, den, agg, tN);
  run_gemm(agg, W1, b1, nullptr, mid, n, d, 2 * d, /*relu=*/1, s);
  run_gemm(mid, W2, b2, /*res=*/h, outC, n, 2 * d, d, /*relu=*/0, s);
}

// ---------------------------------------------------------------------------
// d_in layout (jax sorted-key tree flatten of setup_inputs()):
//  0 mol_x[N,32]  1 conf_x[N,11]  2 conf_edge_attr[E,6]
//  3  conf_dec.edge_lin[128,11]   4  conf_dec.edge_out[11,6]
//  5..10  conf_dec.L0 {W1[11,22],W2[22,11],b1,b2,ln_b,ln_g}
//  11 conf_dec.node_lin[128,11]
//  12 conf_enc.edge_lin[6,128]    13 conf_enc.edge_out[128,128]
//  14..31 conf_enc.L0..L2 (6 each) 32 conf_enc.node_lin[11,128]
//  33 e2d_conf_e  34 e2d_conf_n  35 e2d_mol   (all [128,128])
//  36..41 mol_dec.L0 {W1[32,64],W2[64,32],...}  42 mol_dec.node_lin[128,32]
//  43..60 mol_enc.L0..L2                        61 mol_enc.node_lin[32,128]
//  62 mol_edge_index[2,E] (int32)  63 conf_edge_index[2,E]
// ---------------------------------------------------------------------------
extern "C" void kernel_launch(void* const* d_in, const int* in_sizes, int n_in,
                              void* d_out, int out_size, void* d_ws, size_t ws_size,
                              hipStream_t stream)
{
  if (n_in < 64) return;
  const int IN_DIM = 32, H = 128, CONF_IN = 11, NBINS = 6;
  const int N = in_sizes[0] / IN_DIM;
  const int E = in_sizes[2] / NBINS;

  const float* P[64];
  for (int i = 0; i < 64; ++i) P[i] = (const float*)d_in[i];
  const float* mol_x   = P[0];
  const float* conf_x  = P[1];
  const float* conf_ea = P[2];
  const int* mol_src  = (const int*)d_in[62];
  const int* mol_dst  = mol_src + E;
  const int* conf_src = (const int*)d_in[63];
  const int* conf_dst = conf_src + E;

  // workspace carve (floats)
  float* ws = (float*)d_ws;
  size_t off = 0;
  auto carve = [&](size_t nelem) { float* p = ws + off; off += nelem; return p; };
  float* h_mol  = carve((size_t)N * H);
  float* h_conf = carve((size_t)N * H);
  float* hn     = carve((size_t)N * H);
  float* mid    = carve((size_t)N * 2 * H);
  float* agg    = carve((size_t)N * H);
  float* mmax   = carve((size_t)N * H);
  float* numb   = carve((size_t)N * H);
  float* denb   = carve((size_t)N * H);
  float* m      = carve((size_t)E * H);
  float* e_conf = carve((size_t)E * H);
  float* e2     = carve((size_t)E * H);
  float* xm     = carve((size_t)N * IN_DIM);
  float* eam    = carve((size_t)E * NBINS);
  float* hsm    = carve((size_t)N * IN_DIM);
  (void)ws_size; (void)out_size;

  const unsigned THRESH = 1288490188u;  // 0.3 * 2^32
  hipStream_t s = stream;

  // ---------------- mol encoder (d=128, 3 layers, no edges) ----------------
  {
    long long t = (long long)N * IN_DIM;
    copy_mask_rows<<<(int)((t + 255) / 256), 256, 0, s>>>(mol_x, xm, N, IN_DIM,
                                                          0x1234567u, THRESH);
  }
  run_gemm(xm, P[61], nullptr, nullptr, h_mol, N, IN_DIM, H, 0, s);
  for (int l = 0; l < 3; ++l) {
    int b = 43 + 6 * l;  // W1,W2,b1,b2,ln_b,ln_g
    gen_layer(h_mol, hn, mid, agg, mmax, numb, denb, m,
              mol_src, mol_dst, /*eattr=*/nullptr,
              P[b + 0], P[b + 2], P[b + 1], P[b + 3], P[b + 5], P[b + 4],
              N, E, H, h_mol, s);
  }

  // ---------------- conf encoder (d=128, 3 layers, with edges) -------------
  {
    long long t = (long long)N * CONF_IN;
    copy_mask_rows<<<(int)((t + 255) / 256), 256, 0, s>>>(conf_x, xm, N, CONF_IN,
                                                          0x89ABCDEu, THRESH);
    long long te = (long long)E * NBINS;
    copy_mask_rows<<<(int)((te + 255) / 256), 256, 0, s>>>(conf_ea, eam, E, NBINS,
                                                           0xF00DFACu, THRESH);
  }
  run_gemm(xm,  P[32], nullptr, nullptr, h_conf, N, CONF_IN, H, 0, s);
  run_gemm(eam, P[12], nullptr, nullptr, e_conf, E, NBINS,  H, 0, s);
  for (int l = 0; l < 3; ++l) {
    int b = 14 + 6 * l;
    gen_layer(h_conf, hn, mid, agg, mmax, numb, denb, m,
              conf_src, conf_dst, e_conf,
              P[b + 0], P[b + 2], P[b + 1], P[b + 3], P[b + 5], P[b + 4],
              N, E, H, h_conf, s);
  }
  // conf_edge = e_conf @ conf_enc.edge_out
  run_gemm(e_conf, P[13], nullptr, nullptr, e2, E, H, H, 0, s);

  float* out_mol  = (float*)d_out;                       // [N,32]
  float* out_conf = out_mol + (size_t)N * IN_DIM;        // [N,11]
  float* out_edge = out_conf + (size_t)N * CONF_IN;      // [E,6]

  // ---------------- mol decoder (d=32, 1 layer) ----------------------------
  run_gemm(h_mol, P[35], nullptr, nullptr, mid, N, H, H, 0, s);   // mol_rep
  run_gemm(mid, P[42], nullptr, nullptr, hsm, N, H, IN_DIM, 0, s);
  {
    int b = 36;
    gen_layer(hsm, hn, mid, agg, mmax, numb, denb, m,
              mol_src, mol_dst, /*eattr=*/nullptr,
              P[b + 0], P[b + 2], P[b + 1], P[b + 3], P[b + 5], P[b + 4],
              N, E, IN_DIM, out_mol, s);
  }

  // ---------------- conf decoder (d=11, 1 layer, edges) --------------------
  run_gemm(h_conf, P[34], nullptr, nullptr, mid, N, H, H, 0, s);  // conf_nrep
  run_gemm(mid, P[11], nullptr, nullptr, hsm, N, H, CONF_IN, 0, s);
  run_gemm(e2, P[33], nullptr, nullptr, e_conf, E, H, H, 0, s);   // conf_erep
  run_gemm(e_conf, P[3], nullptr, nullptr, e2, E, H, CONF_IN, 0, s); // e_dec
  {
    int b = 5;
    gen_layer(hsm, hn, mid, agg, mmax, numb, denb, m,
              conf_src, conf_dst, /*eattr=*/e2,
              P[b + 0], P[b + 2], P[b + 1], P[b + 3], P[b + 5], P[b + 4],
              N, E, CONF_IN, out_conf, s);
  }
  // conf_edge_recon = e_dec @ conf_dec.edge_out
  run_gemm(e2, P[4], nullptr, nullptr, out_edge, E, CONF_IN, NBINS, 0, s);
}
